// SwinTransformerBlock_46170898432159
// MI455X (gfx1250) — matrix operand order repaired and checked by reading, not compiled, not run on hardware
//
#include <hip/hip_runtime.h>
#include <hip/hip_bf16.h>
#include <math.h>

typedef __attribute__((ext_vector_type(16))) _Float16 v16h;
typedef __attribute__((ext_vector_type(8)))  float    v8f;

#define WSZ   8
#define NH    8
#define DIM   256
#define HD    32
#define HGT   128
#define WID   128
#define BATCH 8
#define NTOK  (BATCH*HGT*WID)        // 131072
#define NWIN  (BATCH*(HGT/WSZ)*(WID/WSZ))  // 2048
#define ATTN_SCALE 0.17677669529663687f    // 32^-0.5

// ---------------- WMMA fragment loaders (CDNA5 16x16x32 f16 layouts) --------

// A matrix 16x32 f16, row-major in memory (A[m][k], stride lda).
// lanes 0-15: M=lane, K=0..15; lanes 16-31: M=lane-16, K=16..31
__device__ __forceinline__ v16h load_a_frag(const _Float16* __restrict__ A, int lda,
                                            int m0, int k0) {
  int lane = threadIdx.x & 31;
  int row  = m0 + (lane & 15);
  int kb   = (lane >> 4) ? 16 : 0;
  const _Float16* p = A + (size_t)row * lda + k0;
  union { v16h v; uint32_t u[8]; } r;
#pragma unroll
  for (int i = 0; i < 8; ++i) {
    int k = 2 * i + kb;
    r.u[i] = *(const uint32_t*)(p + k);   // packed f16 pair (k, k+1)
  }
  return r.v;
}

// B matrix 32x16 f16 where B[k][n] = Wt[n][k] (Wt row-major [N][K], stride ldk).
// lanes 0-15: K=0..15 ; lanes 16-31: K=16..31 ; VGPR i holds K pair (2i,2i+1)
__device__ __forceinline__ v16h load_b_frag_t(const _Float16* __restrict__ Wt, int ldk,
                                              int k0, int n0) {
  int lane = threadIdx.x & 31;
  int col  = n0 + (lane & 15);
  int kb   = (lane >> 4) ? 16 : 0;
  const _Float16* p = Wt + (size_t)col * ldk + k0 + kb;
  union { v16h v; uint32_t u[8]; } r;
#pragma unroll
  for (int i = 0; i < 8; ++i) r.u[i] = *(const uint32_t*)(p + 2 * i);
  return r.v;
}

// B matrix 32x16 f16 from a row-major [K][N] matrix (stride ldb) — used for V.
__device__ __forceinline__ v16h load_b_frag_rm(const _Float16* __restrict__ B, int ldb,
                                               int k0, int n0) {
  int lane = threadIdx.x & 31;
  int col  = n0 + (lane & 15);
  int kb   = (lane >> 4) ? 16 : 0;
  union { v16h v; _Float16 h[16]; } r;
#pragma unroll
  for (int i = 0; i < 8; ++i) {
    int k = k0 + kb + 2 * i;
    r.h[2 * i]     = B[(size_t)k * ldb + col];
    r.h[2 * i + 1] = B[(size_t)(k + 1) * ldb + col];
  }
  return r.v;
}

// 16x16 output tile over K with V_WMMA_F32_16X16X32_F16
__device__ __forceinline__ v8f wmma_mm(const _Float16* __restrict__ A, int lda,
                                       const _Float16* __restrict__ Wt, int ldk,
                                       int m0, int n0, int K) {
  v8f acc = {};
  for (int k0 = 0; k0 < K; k0 += 32) {
    __builtin_prefetch(A + (size_t)(m0 + (threadIdx.x & 15)) * lda + k0 + 64, 0, 1);
    v16h a  = load_a_frag(A, lda, m0, k0);
    v16h bf = load_b_frag_t(Wt, ldk, k0, n0);
    acc = __builtin_amdgcn_wmma_f32_16x16x32_f16(false, a, false, bf,
                                                 (short)0, acc, false, false);
  }
  return acc;
}

// ---------------- weight convert + transpose to f16 [N][K] ------------------
__global__ void wt_convert_kernel(const float* __restrict__ W, _Float16* __restrict__ Wt,
                                  int K, int N) {
  int i = blockIdx.x * 256 + threadIdx.x;
  if (i >= K * N) return;
  int k = i / N, n = i % N;
  Wt[(size_t)n * K + k] = (_Float16)W[i];
}

// ---------------- LayerNorm -> f16 (1 wave per token) ------------------------
__global__ void ln_f16_kernel(const float* __restrict__ x, const float* __restrict__ g,
                              const float* __restrict__ b, _Float16* __restrict__ out) {
  int t    = blockIdx.x * 8 + (threadIdx.x >> 5);
  int lane = threadIdx.x & 31;
  const float* xp = x + (size_t)t * DIM;
  float v[8]; float s = 0.f;
#pragma unroll
  for (int i = 0; i < 8; ++i) { v[i] = xp[lane + 32 * i]; s += v[i]; }
#pragma unroll
  for (int m = 16; m >= 1; m >>= 1) s += __shfl_xor(s, m, 32);
  float mean = s * (1.0f / DIM);
  float var = 0.f;
#pragma unroll
  for (int i = 0; i < 8; ++i) { float d = v[i] - mean; var += d * d; }
#pragma unroll
  for (int m = 16; m >= 1; m >>= 1) var += __shfl_xor(var, m, 32);
  float rstd = rsqrtf(var * (1.0f / DIM) + 1e-5f);
  _Float16* op = out + (size_t)t * DIM;
#pragma unroll
  for (int i = 0; i < 8; ++i) {
    int c = lane + 32 * i;
    op[c] = (_Float16)((v[i] - mean) * rstd * g[c] + b[c]);
  }
}

// ---------------- QKV GEMM: [NTOK,256]x[256,768], permuted windowed store ----
__global__ void gemm_qkv_kernel(const _Float16* __restrict__ X,
                                const _Float16* __restrict__ Wt,
                                const float* __restrict__ bias,
                                _Float16* __restrict__ qkv) {
  const int Ntiles = 768 / 16;
  int tile = blockIdx.x * 8 + (threadIdx.x >> 5);
  int mt = tile / Ntiles, nt = tile % Ntiles;
  int m0 = mt * 16, n0 = nt * 16;
  v8f acc = wmma_mm(X, DIM, Wt, DIM, m0, n0, DIM);
  int lane = threadIdx.x & 31;
  int col  = n0 + (lane & 15);
  int hi   = (lane >> 4) ? 8 : 0;
  float bv = bias[col];
#pragma unroll
  for (int r = 0; r < 8; ++r) {
    int m = m0 + r + hi;                 // source token index
    int b = m >> 14, rem = m & 16383;
    int h = rem >> 7, w = rem & 127;
    int hs  = (h + 124) & 127;           // roll(-4)
    int wsx = (w + 124) & 127;
    int win = ((hs >> 3) << 4) | (wsx >> 3);
    int g   = (((b << 8) | win) << 6) | (((hs & 7) << 3) | (wsx & 7));
    qkv[(size_t)g * 768 + col] = (_Float16)(acc[r] + bv);
  }
}

// ---------------- windowed attention: 1 wave per (window, head) --------------
__global__ __launch_bounds__(32) void attn_kernel(const _Float16* __restrict__ qkv,
                                                  const float* __restrict__ rel_bias,
                                                  _Float16* __restrict__ o16) {
  __shared__ float    S[64 * 64];
  __shared__ _Float16 P[64 * 64];
  int head = blockIdx.x & 7;
  int win  = blockIdx.x >> 3;
  const _Float16* base = qkv + (size_t)win * 64 * 768;
  const _Float16* Qb = base + head * HD;
  const _Float16* Kb = base + 256 + head * HD;
  const _Float16* Vb = base + 512 + head * HD;
  int lane = threadIdx.x & 31;
  int hi   = (lane >> 4) ? 8 : 0;
  int cn   = lane & 15;

  // S = Q * K^T  (64x64, K-dim = 32 -> single WMMA per tile)
  v16h qa[4], kb[4];
#pragma unroll
  for (int i = 0; i < 4; ++i) qa[i] = load_a_frag(Qb, 768, i * 16, 0);
#pragma unroll
  for (int i = 0; i < 4; ++i) kb[i] = load_b_frag_t(Kb, 768, 0, i * 16);
#pragma unroll
  for (int mt = 0; mt < 4; ++mt)
#pragma unroll
    for (int nt = 0; nt < 4; ++nt) {
      v8f acc = {};
      acc = __builtin_amdgcn_wmma_f32_16x16x32_f16(false, qa[mt], false, kb[nt],
                                                   (short)0, acc, false, false);
#pragma unroll
      for (int r = 0; r < 8; ++r)
        S[(mt * 16 + r + hi) * 64 + nt * 16 + cn] = acc[r];
    }
  __syncthreads();

  // softmax rows (2 rows per lane), relative-position bias computed in place
#pragma unroll
  for (int rr = 0; rr < 2; ++rr) {
    int m  = lane * 2 + rr;
    int mh = m >> 3, mw = m & 7;
    float mx = -1e30f;
    for (int n = 0; n < 64; ++n) {
      int dh = mh - (n >> 3) + 7, dw = mw - (n & 7) + 7;
      float vv = S[m * 64 + n] * ATTN_SCALE + rel_bias[(dh * 15 + dw) * NH + head];
      S[m * 64 + n] = vv;
      mx = fmaxf(mx, vv);
    }
    float sum = 0.f;
    for (int n = 0; n < 64; ++n) {
      float e = __expf(S[m * 64 + n] - mx);
      S[m * 64 + n] = e; sum += e;
    }
    float inv = 1.0f / sum;
    for (int n = 0; n < 64; ++n) P[m * 64 + n] = (_Float16)(S[m * 64 + n] * inv);
  }
  __syncthreads();

  // O = P * V  (64x32)
  v16h vb[2][2];
#pragma unroll
  for (int ks = 0; ks < 2; ++ks)
#pragma unroll
    for (int nt = 0; nt < 2; ++nt) vb[ks][nt] = load_b_frag_rm(Vb, 768, ks * 32, nt * 16);
#pragma unroll
  for (int mt = 0; mt < 4; ++mt) {
    v16h a0 = load_a_frag(P, 64, mt * 16, 0);
    v16h a1 = load_a_frag(P, 64, mt * 16, 32);
#pragma unroll
    for (int nt = 0; nt < 2; ++nt) {
      v8f acc = {};
      acc = __builtin_amdgcn_wmma_f32_16x16x32_f16(false, a0, false, vb[0][nt],
                                                   (short)0, acc, false, false);
      acc = __builtin_amdgcn_wmma_f32_16x16x32_f16(false, a1, false, vb[1][nt],
                                                   (short)0, acc, false, false);
#pragma unroll
      for (int r = 0; r < 8; ++r) {
        int m = mt * 16 + r + hi;
        o16[(size_t)(win * 64 + m) * DIM + head * HD + nt * 16 + cn] = (_Float16)acc[r];
      }
    }
  }
}

// ---------------- proj GEMM + un-shift + residual -> f32 x1 ------------------
__global__ void gemm_proj_kernel(const _Float16* __restrict__ O,
                                 const _Float16* __restrict__ Wt,
                                 const float* __restrict__ bias,
                                 const float* __restrict__ x,
                                 float* __restrict__ x1) {
  const int Ntiles = DIM / 16;
  int tile = blockIdx.x * 8 + (threadIdx.x >> 5);
  int mt = tile / Ntiles, nt = tile % Ntiles;
  int m0 = mt * 16, n0 = nt * 16;
  v8f acc = wmma_mm(O, DIM, Wt, DIM, m0, n0, DIM);
  int lane = threadIdx.x & 31;
  int col  = n0 + (lane & 15);
  int hi   = (lane >> 4) ? 8 : 0;
  float bv = bias[col];
#pragma unroll
  for (int r = 0; r < 8; ++r) {
    int g = m0 + r + hi;                 // windowed row -> token index
    int wing = g >> 6, pos = g & 63;
    int b = wing >> 8, wloc = wing & 255;
    int hs  = ((wloc >> 4) << 3) | (pos >> 3);
    int wsx = ((wloc & 15) << 3) | (pos & 7);
    int h = (hs + 4) & 127, w = (wsx + 4) & 127;   // roll(+4)
    size_t t = ((((size_t)b << 7) | h) << 7) | w;
    x1[t * DIM + col] = x[t * DIM + col] + acc[r] + bv;
  }
}

// ---------------- fc1 GEMM + exact GELU -> f16 -------------------------------
__global__ void gemm_fc1_kernel(const _Float16* __restrict__ H2,
                                const _Float16* __restrict__ Wt,
                                const float* __restrict__ bias,
                                _Float16* __restrict__ M1) {
  const int Ntiles = DIM / 16;
  int tile = blockIdx.x * 8 + (threadIdx.x >> 5);
  int mt = tile / Ntiles, nt = tile % Ntiles;
  int m0 = mt * 16, n0 = nt * 16;
  v8f acc = wmma_mm(H2, DIM, Wt, DIM, m0, n0, DIM);
  int lane = threadIdx.x & 31;
  int col  = n0 + (lane & 15);
  int hi   = (lane >> 4) ? 8 : 0;
  float bv = bias[col];
#pragma unroll
  for (int r = 0; r < 8; ++r) {
    float v = acc[r] + bv;
    float ge = 0.5f * v * (1.0f + erff(v * 0.70710678118654752f));
    M1[(size_t)(m0 + r + hi) * DIM + col] = (_Float16)ge;
  }
}

// ---------------- fc2 GEMM + residual -> f32 output --------------------------
__global__ void gemm_fc2_kernel(const _Float16* __restrict__ M1,
                                const _Float16* __restrict__ Wt,
                                const float* __restrict__ bias,
                                const float* __restrict__ x1,
                                float* __restrict__ out) {
  const int Ntiles = DIM / 16;
  int tile = blockIdx.x * 8 + (threadIdx.x >> 5);
  int mt = tile / Ntiles, nt = tile % Ntiles;
  int m0 = mt * 16, n0 = nt * 16;
  v8f acc = wmma_mm(M1, DIM, Wt, DIM, m0, n0, DIM);
  int lane = threadIdx.x & 31;
  int col  = n0 + (lane & 15);
  int hi   = (lane >> 4) ? 8 : 0;
  float bv = bias[col];
#pragma unroll
  for (int r = 0; r < 8; ++r) {
    size_t t = (size_t)(m0 + r + hi);
    out[t * DIM + col] = x1[t * DIM + col] + acc[r] + bv;
  }
}

// ---------------- host launcher ----------------------------------------------
extern "C" void kernel_launch(void* const* d_in, const int* in_sizes, int n_in,
                              void* d_out, int out_size, void* d_ws, size_t ws_size,
                              hipStream_t stream) {
  const float* x       = (const float*)d_in[0];
  const float* n1g     = (const float*)d_in[1];
  const float* n1b     = (const float*)d_in[2];
  const float* qkv_w   = (const float*)d_in[3];
  const float* qkv_b   = (const float*)d_in[4];
  const float* relb    = (const float*)d_in[5];
  const float* proj_w  = (const float*)d_in[6];
  const float* proj_b  = (const float*)d_in[7];
  const float* n2g     = (const float*)d_in[8];
  const float* n2b     = (const float*)d_in[9];
  const float* fc1_w   = (const float*)d_in[10];
  const float* fc1_b   = (const float*)d_in[11];
  const float* fc2_w   = (const float*)d_in[12];
  const float* fc2_b   = (const float*)d_in[13];
  float* outp = (float*)d_out;

  // workspace layout (bytes, all 256-aligned)
  char* ws = (char*)d_ws;
  _Float16* wt_qkv  = (_Float16*)(ws);                               // 768*256*2
  _Float16* wt_proj = (_Float16*)(ws + 393216);                      // 256*256*2
  _Float16* wt_fc1  = (_Float16*)(ws + 524288);
  _Float16* wt_fc2  = (_Float16*)(ws + 655360);
  _Float16* x16     = (_Float16*)(ws + 786432);                      // 131072*256*2 (reused as H2)
  _Float16* o16     = (_Float16*)(ws + 786432 + 67108864ull);        // 131072*256*2 (reused as M1)
  _Float16* qkv16   = (_Float16*)(ws + 786432 + 134217728ull);       // 131072*768*2
  float*    x1      = (float*)   (ws + 786432 + 134217728ull + 201326592ull); // 131072*256*4

  // 1) weights -> f16 transposed [N][K]
  wt_convert_kernel<<<(256 * 768 + 255) / 256, 256, 0, stream>>>(qkv_w,  wt_qkv,  256, 768);
  wt_convert_kernel<<<(256 * 256 + 255) / 256, 256, 0, stream>>>(proj_w, wt_proj, 256, 256);
  wt_convert_kernel<<<(256 * 256 + 255) / 256, 256, 0, stream>>>(fc1_w,  wt_fc1,  256, 256);
  wt_convert_kernel<<<(256 * 256 + 255) / 256, 256, 0, stream>>>(fc2_w,  wt_fc2,  256, 256);

  // 2) LN1 -> f16
  ln_f16_kernel<<<NTOK / 8, 256, 0, stream>>>(x, n1g, n1b, x16);

  // 3) QKV GEMM with windowed permute store
  gemm_qkv_kernel<<<(NTOK / 16) * (768 / 16) / 8, 256, 0, stream>>>(x16, wt_qkv, qkv_b, qkv16);

  // 4) windowed attention (1 wave per window-head)
  attn_kernel<<<NWIN * NH, 32, 0, stream>>>(qkv16, relb, o16);

  // 5) proj GEMM + un-shift + residual
  gemm_proj_kernel<<<(NTOK / 16) * (DIM / 16) / 8, 256, 0, stream>>>(o16, wt_proj, proj_b, x, x1);

  // 6) LN2 -> f16 (reuse x16)
  ln_f16_kernel<<<NTOK / 8, 256, 0, stream>>>(x1, n2g, n2b, x16);

  // 7) fc1 + GELU -> f16 (reuse o16)
  gemm_fc1_kernel<<<(NTOK / 16) * (DIM / 16) / 8, 256, 0, stream>>>(x16, wt_fc1, fc1_b, o16);

  // 8) fc2 + residual -> f32 output
  gemm_fc2_kernel<<<(NTOK / 16) * (DIM / 16) / 8, 256, 0, stream>>>(o16, wt_fc2, fc2_b, x1, outp);
}
